// LSTMModel_17274358464706
// MI455X (gfx1250) — matrix operand order repaired and checked by reading, not compiled, not run on hardware
//
#include <hip/hip_runtime.h>
#include <hip/hip_bf16.h>

namespace {

constexpr int kB   = 1024;
constexpr int kT   = 20;
constexpr int kDin = 4;
constexpr int kH   = 768;
constexpr int kG   = 4 * kH;   // 3072

typedef __bf16 bf16_t;
typedef bf16_t v16bf __attribute__((ext_vector_type(16)));
typedef bf16_t v8bf  __attribute__((ext_vector_type(8)));
typedef bf16_t v4bf  __attribute__((ext_vector_type(4)));
typedef float  v8f   __attribute__((ext_vector_type(8)));

__device__ __forceinline__ float sigm(float x) { return 1.0f / (1.0f + __expf(-x)); }

// ---------------------------------------------------------------------------
// Kernel 1: rowsum over W_hh rows: rs[t*kG + j] = sum_k W_hh[t][j][k]
// One wave32 per row; float4 coalesced loads; shfl tree reduce (deterministic).
// ---------------------------------------------------------------------------
__global__ __launch_bounds__(256) void rowsum_kernel(const float* __restrict__ whh,
                                                     float* __restrict__ rs) {
  const int wave = (int)((blockIdx.x * blockDim.x + threadIdx.x) >> 5);
  const int lane = (int)(threadIdx.x & 31);
  if (wave >= kT * kG) return;
  const float4* row = (const float4*)(whh + (size_t)wave * kH);
  float s = 0.0f;
#pragma unroll
  for (int i = 0; i < kH / 128; ++i) {   // 6 float4 per lane
    float4 v = row[i * 32 + lane];
    s += (v.x + v.y) + (v.z + v.w);
  }
#pragma unroll
  for (int off = 16; off > 0; off >>= 1) s += __shfl_xor(s, off, 32);
  if (lane == 0) rs[wave] = s;
}

// ---------------------------------------------------------------------------
// Kernel 1b: one-shot f32 -> bf16 staging of rows (row-major dst, width kH).
// Used for W_hh[0] (3072 rows, stride kH) and ev0 (1024 rows, stride kT*kH).
// These arrays are re-read dozens of times by the WMMA GEMM, so converting
// once halves the GEMM read bytes and removes all cvts from its inner loop.
// ---------------------------------------------------------------------------
__global__ __launch_bounds__(256) void cvt_rows_kernel(const float* __restrict__ src,
                                                       bf16_t* __restrict__ dst,
                                                       int rows, int src_stride) {
  const int idx = (int)blockIdx.x * 256 + (int)threadIdx.x;  // one float4 each
  const int per_row = kH / 4;                                // 192
  const int r = idx / per_row;
  const int q = (idx % per_row) * 4;
  if (r >= rows) return;
  const float4 v = *(const float4*)(src + (size_t)r * src_stride + q);
  v4bf o;
  o[0] = (bf16_t)v.x; o[1] = (bf16_t)v.y; o[2] = (bf16_t)v.z; o[3] = (bf16_t)v.w;
  *(v4bf*)(dst + (size_t)r * kH + q) = o;
}

// ---------------------------------------------------------------------------
// Kernel 2: step 0 — gates = ev0 @ W_hh[0]^T (+ x0 @ W_ih[0]^T + b) via
// v_wmma_f32_16x16x32_bf16 on pre-staged bf16 operands. One wave per 16x16
// (b,k) tile; 4 gate accumulators share the A tile. K-loop is software-
// pipelined (ping/pong register buffers) so loads overlap the WMMAs.
// Fused LSTM pointwise -> h0, c0, plus per-tile partial of (h0+ev1) for hx_1.
// ---------------------------------------------------------------------------
__global__ __launch_bounds__(32) void step0_kernel(
    const float* __restrict__ price, const float* __restrict__ event_,
    const bf16_t* __restrict__ ev_bf, const bf16_t* __restrict__ whh_bf,
    const float* __restrict__ wih0,
    const float* __restrict__ bih0, const float* __restrict__ bhh0,
    float* __restrict__ h, float* __restrict__ c, float* __restrict__ partial) {
  const int bi = (int)blockIdx.x;           // 0..63  (b tiles)
  const int ki = (int)blockIdx.y;           // 0..47  (k tiles)
  const int lane = (int)threadIdx.x;        // 0..31
  const int half = lane >> 4;
  const int mn   = lane & 15;
  const int b_base = bi * 16;
  const int k_base = ki * 16;

  v8f zero = {0.f, 0.f, 0.f, 0.f, 0.f, 0.f, 0.f, 0.f};
  v8f acc[4] = {zero, zero, zero, zero};

  // A 16x32 bf16 fragment: lane(m=mn, half); elem e -> K=kt+16*half+e
  // => two adjacent 16B chunks.
  const bf16_t* ap = ev_bf + (size_t)(b_base + mn) * kH + half * 16;
  // B 32x16 bf16 fragment: lane(n=mn, half); elem e -> K=kt+16*half+e
  // => one contiguous 32B chunk from W_hh row (g*kH + k_base + n).
  const bf16_t* bp[4];
#pragma unroll
  for (int g = 0; g < 4; ++g)
    bp[g] = whh_bf + (size_t)(g * kH + k_base + mn) * kH + half * 16;

  auto load_tile = [&](int kt, v16bf& a, v16bf* bm) {
    union { v16bf v; v8bf hv[2]; } u;
    u.hv[0] = *(const v8bf*)(ap + kt);
    u.hv[1] = *(const v8bf*)(ap + kt + 8);
    a = u.v;
#pragma unroll
    for (int g = 0; g < 4; ++g) bm[g] = *(const v16bf*)(bp[g] + kt);
  };

  v16bf a0, a1, b0[4], b1[4];
  load_tile(0, a0, b0);
#pragma unroll
  for (int kt = 0; kt < kH; kt += 64) {
    load_tile(kt + 32, a1, b1);           // in flight during first 4 WMMAs
#pragma unroll
    for (int g = 0; g < 4; ++g)
      acc[g] = __builtin_amdgcn_wmma_f32_16x16x32_bf16(
          false, a0, false, b0[g], (short)0, acc[g], false, false);
    if (kt + 64 < kH) load_tile(kt + 64, a0, b0);  // in flight during next 4
#pragma unroll
    for (int g = 0; g < 4; ++g)
      acc[g] = __builtin_amdgcn_wmma_f32_16x16x32_bf16(
          false, a1, false, b1[g], (short)0, acc[g], false, false);
  }

  // C/D layout: lane -> n = mn; VGPR r -> m = 8*half + r.
  const int col_k = k_base + mn;
  float4 wi[4];
  float  bsum[4];
#pragma unroll
  for (int g = 0; g < 4; ++g) {
    const int col = g * kH + col_k;
    wi[g]   = *(const float4*)(wih0 + (size_t)col * kDin);
    bsum[g] = bih0[col] + bhh0[col];
  }

  float psum = 0.0f;
#pragma unroll
  for (int r = 0; r < 8; ++r) {
    const int b = b_base + half * 8 + r;
    const float4 xv = *(const float4*)(price + (size_t)b * (kT * kDin));  // t=0
    float gate[4];
#pragma unroll
    for (int g = 0; g < 4; ++g)
      gate[g] = acc[g][r] + xv.x * wi[g].x + xv.y * wi[g].y +
                xv.z * wi[g].z + xv.w * wi[g].w + bsum[g];
    // c_prev == 0  ->  c0 = sig(i)*tanh(g);  h0 = sig(o)*tanh(c0)
    const float cn = sigm(gate[0]) * tanhf(gate[2]);
    const float hn = sigm(gate[3]) * tanhf(cn);
    c[(size_t)b * kH + col_k] = cn;
    h[(size_t)b * kH + col_k] = hn;
    psum += hn + event_[(size_t)b * (kT * kH) + kH + col_k];  // + ev at t=1
  }
#pragma unroll
  for (int off = 16; off > 0; off >>= 1) psum += __shfl_xor(psum, off, 32);
  if (lane == 0) partial[ki * (kB / 16) + bi] = psum;
}

// ---------------------------------------------------------------------------
// Kernel 3: recurrent step t in 1..19.
//   hx = sum(partials_in)        (every block reduces; broadcast via LDS)
//   gates[b][g*H+k] = x[b]·W_ih[t][col] + b_ih + b_hh + hx * rowsum[col]
//   c,h update; emit per-block partial of (h_t + ev_{t+1}) for the next step.
// ---------------------------------------------------------------------------
__global__ __launch_bounds__(256) void step_kernel(
    int t, const float* __restrict__ price, const float* __restrict__ event_,
    const float* __restrict__ wih, const float* __restrict__ bih,
    const float* __restrict__ bhh, const float* __restrict__ rs,
    float* __restrict__ h, float* __restrict__ c,
    const float* __restrict__ pin, int npin, float* __restrict__ pout,
    int produce) {
  __shared__ float sred[256];
  const int tid = (int)threadIdx.x;

  float s = 0.0f;
  for (int i = tid; i < npin; i += 256) s += pin[i];
  sred[tid] = s;
  __syncthreads();
#pragma unroll
  for (int off = 128; off > 0; off >>= 1) {
    if (tid < off) sred[tid] += sred[tid + off];
    __syncthreads();
  }
  const float hx = sred[0];
  __syncthreads();

  const int idx = (int)blockIdx.x * 256 + tid;
  const int k4  = idx % (kH / 4);
  const int b   = idx / (kH / 4);
  const int k   = k4 * 4;

  const float4 xv  = *(const float4*)(price + ((size_t)b * kT + t) * kDin);
  const float* wt  = wih + (size_t)t * kG * kDin;
  const float* bit = bih + (size_t)t * kG;
  const float* bht = bhh + (size_t)t * kG;
  const float* rst = rs  + (size_t)t * kG;

  const float4 cold = *(const float4*)(c + (size_t)b * kH + k);
  const float co[4] = {cold.x, cold.y, cold.z, cold.w};
  float cn[4], hn[4];
#pragma unroll
  for (int kk = 0; kk < 4; ++kk) {
    const int kc = k + kk;
    float gate[4];
#pragma unroll
    for (int g = 0; g < 4; ++g) {
      const int col = g * kH + kc;
      const float4 w = *(const float4*)(wt + (size_t)col * kDin);
      gate[g] = xv.x * w.x + xv.y * w.y + xv.z * w.z + xv.w * w.w +
                bit[col] + bht[col] + hx * rst[col];
    }
    cn[kk] = sigm(gate[1]) * co[kk] + sigm(gate[0]) * tanhf(gate[2]);
    hn[kk] = sigm(gate[3]) * tanhf(cn[kk]);
  }
  const float4 cv = {cn[0], cn[1], cn[2], cn[3]};
  const float4 hv = {hn[0], hn[1], hn[2], hn[3]};
  *(float4*)(c + (size_t)b * kH + k) = cv;
  *(float4*)(h + (size_t)b * kH + k) = hv;

  float ps = 0.0f;
  if (produce) {
    const float4 ev = *(const float4*)(event_ + ((size_t)b * kT + (t + 1)) * kH + k);
    ps = (hn[0] + hn[1]) + (hn[2] + hn[3]) + (ev.x + ev.y) + (ev.z + ev.w);
  }
  sred[tid] = ps;
  __syncthreads();
#pragma unroll
  for (int off = 128; off > 0; off >>= 1) {
    if (tid < off) sred[tid] += sred[tid + off];
    __syncthreads();
  }
  if (tid == 0 && produce) pout[blockIdx.x] = sred[0];
}

// ---------------------------------------------------------------------------
// Kernel 4: out[b] = h19[b] · fc_w + fc_b     (one wave per batch row)
// ---------------------------------------------------------------------------
__global__ __launch_bounds__(256) void fc_kernel(const float* __restrict__ h,
                                                 const float* __restrict__ fcw,
                                                 const float* __restrict__ fcb,
                                                 float* __restrict__ out) {
  const int wave = (int)((blockIdx.x * blockDim.x + threadIdx.x) >> 5);
  const int lane = (int)(threadIdx.x & 31);
  if (wave >= kB) return;
  const float* row = h + (size_t)wave * kH;
  float s = 0.0f;
#pragma unroll
  for (int i = 0; i < kH / 32; ++i) s += row[i * 32 + lane] * fcw[i * 32 + lane];
#pragma unroll
  for (int off = 16; off > 0; off >>= 1) s += __shfl_xor(s, off, 32);
  if (lane == 0) out[wave] = s + fcb[0];
}

}  // namespace

extern "C" void kernel_launch(void* const* d_in, const int* in_sizes, int n_in,
                              void* d_out, int out_size, void* d_ws, size_t ws_size,
                              hipStream_t stream) {
  (void)in_sizes; (void)n_in; (void)out_size; (void)ws_size;
  const float* price  = (const float*)d_in[0];  // (B,T,4)
  const float* event_ = (const float*)d_in[1];  // (B,T,H)
  const float* Wih    = (const float*)d_in[2];  // (T,4H,4)
  const float* Whh    = (const float*)d_in[3];  // (T,4H,H)
  const float* bih    = (const float*)d_in[4];  // (T,4H)
  const float* bhh    = (const float*)d_in[5];  // (T,4H)
  const float* fcw    = (const float*)d_in[6];  // (1,H)
  const float* fcb    = (const float*)d_in[7];  // (1,)
  float* out = (float*)d_out;

  float* rs = (float*)d_ws;                        // kT*kG floats (240 KB)
  float* h  = rs + (size_t)kT * kG;                // B*H floats (3 MB)
  float* c  = h + (size_t)kB * kH;                 // B*H floats (3 MB)
  float* pa = c + (size_t)kB * kH;                 // 3072 partials
  float* pb = pa + (kB / 16) * (kH / 16);          // 768-wide second buffer
  bf16_t* whh_bf = (bf16_t*)(pb + 768);            // 3072*768 bf16 (4.5 MB)
  bf16_t* ev_bf  = whh_bf + (size_t)kG * kH;       // 1024*768 bf16 (1.5 MB)

  // 1) rowsums of W_hh (streams 189 MB — the HBM-bound bulk of the problem)
  rowsum_kernel<<<(kT * kG) / 8, 256, 0, stream>>>(Whh, rs);

  // 1b) one-shot bf16 staging of the step-0 GEMM operands
  cvt_rows_kernel<<<(kG * (kH / 4)) / 256, 256, 0, stream>>>(Whh, whh_bf, kG, kH);
  cvt_rows_kernel<<<(kB * (kH / 4)) / 256, 256, 0, stream>>>(event_, ev_bf, kB,
                                                             kT * kH);

  // 2) step 0: pipelined WMMA GEMM + fused LSTM pointwise + partials for hx_1
  step0_kernel<<<dim3(kB / 16, kH / 16), 32, 0, stream>>>(
      price, event_, ev_bf, whh_bf, Wih, bih, bhh, h, c, pa);

  // 3) steps 1..19: one fused kernel per step, ping-pong partial buffers
  const float* pin = pa;
  float* pout = pb;
  int npin = (kB / 16) * (kH / 16);  // 3072 partials from step0
  for (int t = 1; t < kT; ++t) {
    const int blocks = (kB * kH / 4) / 256;  // 768
    step_kernel<<<blocks, 256, 0, stream>>>(t, price, event_, Wih, bih, bhh, rs,
                                            h, c, pin, npin, pout,
                                            (t < kT - 1) ? 1 : 0);
    const float* tmp = pout; pout = (float*)pin; pin = tmp;
    npin = blocks;
  }

  // 4) final FC: out = h19 @ fc_w^T + fc_b
  fc_kernel<<<(kB * 32) / 256, 256, 0, stream>>>(h, fcw, fcb, out);
}
